// EncoderLoc_45835890983481
// MI455X (gfx1250) — compile-verified, matrix operand order checked by reading, not run on hardware
//
#include <hip/hip_runtime.h>

#define B_   4
#define N1_  16384
#define S1_  2048
#define S2_  1024
#define KNB_ 32
#define EPSN 1e-5f

typedef __attribute__((ext_vector_type(2)))  float    v2f;
typedef __attribute__((ext_vector_type(8)))  float    v8f;
typedef __attribute__((ext_vector_type(16))) _Float16 v16h;

// ---------------------------------------------------------------------------
// Input transpose: pc/feature [B,3,N] -> xyz/feat [B,N,3]; also pass pc -> out
// ---------------------------------------------------------------------------
__global__ void k_transpose_in(const float* __restrict__ pc, const float* __restrict__ feature,
                               float* __restrict__ xyz, float* __restrict__ feat,
                               float* __restrict__ out_pc, int Bn, int N) {
    int total = Bn * N;
    for (int p = blockIdx.x * blockDim.x + threadIdx.x; p < total; p += gridDim.x * blockDim.x) {
        int b = p / N, n = p % N;
#pragma unroll
        for (int c = 0; c < 3; ++c) {
            float v = pc[((size_t)b * 3 + c) * N + n];
            xyz[(size_t)p * 3 + c] = v;
            out_pc[((size_t)b * 3 + c) * N + n] = v;
            feat[(size_t)p * 3 + c] = feature[((size_t)b * 3 + c) * N + n];
        }
    }
}

// ---------------------------------------------------------------------------
// Farthest point sampling. One workgroup per batch; whole point cloud lives in
// LDS (CDNA5: 320 KB/WGP). dist[] kept in registers (PP points per thread).
// wave32 shuffle reductions, tie-break = smallest index (matches jnp.argmax).
// ---------------------------------------------------------------------------
template <int PP>
__global__ void k_fps(const float* __restrict__ xyz, int* __restrict__ fidx,
                      int* __restrict__ fidx_out, int N, int S) {
    extern __shared__ float sm[];
    float* sx = sm;
    float* sy = sm + N;
    float* sz = sm + 2 * N;
    float* rv = sm + 3 * N;          // 32 floats
    int*   ri = (int*)(rv + 32);     // 32 ints
    int*   sfar = ri + 32;           // 1 int

    int b = blockIdx.x, t = threadIdx.x, T = blockDim.x;
    const float* base = xyz + (size_t)b * N * 3;
    for (int i = t; i < N; i += T) {
        sx[i] = base[(size_t)i * 3 + 0];
        sy[i] = base[(size_t)i * 3 + 1];
        sz[i] = base[(size_t)i * 3 + 2];
    }
    __syncthreads();

    float dist[PP];
#pragma unroll
    for (int j = 0; j < PP; ++j) dist[j] = 1e10f;
    int far = 0;

    for (int i = 0; i < S; ++i) {
        if (t == 0) { fidx[b * S + i] = far; fidx_out[b * S + i] = far; }
        float cx = sx[far], cy = sy[far], cz = sz[far];
        float best = -1.0f; int bi = 0x7fffffff;
#pragma unroll
        for (int j = 0; j < PP; ++j) {
            int n = t + j * T;
            float dx = sx[n] - cx, dy = sy[n] - cy, dz = sz[n] - cz;
            float d = dx * dx + dy * dy + dz * dz;
            dist[j] = fminf(dist[j], d);
            if (dist[j] > best || (dist[j] == best && n < bi)) { best = dist[j]; bi = n; }
        }
        // wave32 reduce
#pragma unroll
        for (int offl = 16; offl > 0; offl >>= 1) {
            float ov = __shfl_down(best, offl);
            int   oi = __shfl_down(bi, offl);
            if (ov > best || (ov == best && oi < bi)) { best = ov; bi = oi; }
        }
        int wid = t >> 5;
        if ((t & 31) == 0) { rv[wid] = best; ri[wid] = bi; }
        __syncthreads();
        if (t < 32) {
            int nw = T >> 5;
            float v2 = (t < nw) ? rv[t] : -1.0f;
            int   i2 = (t < nw) ? ri[t] : 0x7fffffff;
#pragma unroll
            for (int offl = 16; offl > 0; offl >>= 1) {
                float ov = __shfl_down(v2, offl);
                int   oi = __shfl_down(i2, offl);
                if (ov > v2 || (ov == v2 && oi < i2)) { v2 = ov; i2 = oi; }
            }
            if (t == 0) *sfar = i2;
        }
        __syncthreads();
        far = *sfar;
    }
}

// ---------------------------------------------------------------------------
// Gather sampled centers; also emit channel-first pc_lX output [B,3,S]
// ---------------------------------------------------------------------------
__global__ void k_gather_centers(const float* __restrict__ xyz, const int* __restrict__ fidx,
                                 float* __restrict__ nx, float* __restrict__ out_pcl,
                                 int N, int S) {
    int total = B_ * S;
    for (int p = blockIdx.x * blockDim.x + threadIdx.x; p < total; p += gridDim.x * blockDim.x) {
        int b = p / S, s = p % S;
        int id = fidx[p];
#pragma unroll
        for (int c = 0; c < 3; ++c) {
            float v = xyz[((size_t)b * N + id) * 3 + c];
            nx[(size_t)p * 3 + c] = v;
            out_pcl[((size_t)b * 3 + c) * S + s] = v;
        }
    }
}

// ---------------------------------------------------------------------------
// Brute-force kNN (K=32). One thread per query; refs tiled through LDS;
// register-resident sorted top-32 with fully unrolled predicated insertion.
// ---------------------------------------------------------------------------
#define KNN_TILE 512
__global__ void k_knn(const float* __restrict__ query, const float* __restrict__ ref,
                      int* __restrict__ nidx, int N, int S) {
    __shared__ float tx[KNN_TILE], ty[KNN_TILE], tz[KNN_TILE];
    int gq = blockIdx.x * blockDim.x + threadIdx.x;   // S % blockDim == 0, no straddle
    int b = gq / S;
    float qx = query[(size_t)gq * 3 + 0];
    float qy = query[(size_t)gq * 3 + 1];
    float qz = query[(size_t)gq * 3 + 2];
    const float* rbase = ref + (size_t)b * N * 3;

    float nd[KNB_]; int ni[KNB_];
#pragma unroll
    for (int j = 0; j < KNB_; ++j) { nd[j] = 3.4e38f; ni[j] = 0; }

    for (int t0 = 0; t0 < N; t0 += KNN_TILE) {
        for (int i = threadIdx.x; i < KNN_TILE; i += blockDim.x) {
            __builtin_prefetch(&rbase[(size_t)(t0 + i + KNN_TILE) * 3], 0, 0);
            tx[i] = rbase[(size_t)(t0 + i) * 3 + 0];
            ty[i] = rbase[(size_t)(t0 + i) * 3 + 1];
            tz[i] = rbase[(size_t)(t0 + i) * 3 + 2];
        }
        __syncthreads();
        for (int i = 0; i < KNN_TILE; ++i) {
            float dx = qx - tx[i], dy = qy - ty[i], dz = qz - tz[i];
            float d = dx * dx + dy * dy + dz * dz;
            if (d < nd[KNB_ - 1]) {
                nd[KNB_ - 1] = d; ni[KNB_ - 1] = t0 + i;
#pragma unroll
                for (int j = KNB_ - 1; j > 0; --j) {
                    bool sw = nd[j] < nd[j - 1];
                    float fa = sw ? nd[j] : nd[j - 1];
                    float fb = sw ? nd[j - 1] : nd[j];
                    int   ia = sw ? ni[j] : ni[j - 1];
                    int   ib = sw ? ni[j - 1] : ni[j];
                    nd[j - 1] = fa; nd[j] = fb; ni[j - 1] = ia; ni[j] = ib;
                }
            }
        }
        __syncthreads();
    }
#pragma unroll
    for (int j = 0; j < KNB_; ++j) nidx[(size_t)gq * KNB_ + j] = ni[j];
}

// ---------------------------------------------------------------------------
// Grouping: build X rows = [rel_xyz | feat | zero-pad]
// ---------------------------------------------------------------------------
__global__ void k_group_sa1(const float* __restrict__ xyz, const float* __restrict__ feat,
                            const float* __restrict__ nx, const int* __restrict__ nidx,
                            float* __restrict__ X, int N, int S) {
    int total = B_ * S * KNB_;
    for (int p = blockIdx.x * blockDim.x + threadIdx.x; p < total; p += gridDim.x * blockDim.x) {
        int b = p / (S * KNB_);
        int rem = p % (S * KNB_);
        int s = rem / KNB_;
        int id = nidx[p];
        const float* src = xyz + ((size_t)b * N + id) * 3;
        const float* ctr = nx + ((size_t)b * S + s) * 3;
        const float* f = feat + ((size_t)b * N + id) * 3;
        float* row = X + (size_t)p * 8;
        row[0] = src[0] - ctr[0]; row[1] = src[1] - ctr[1]; row[2] = src[2] - ctr[2];
        row[3] = f[0]; row[4] = f[1]; row[5] = f[2];
        row[6] = 0.0f; row[7] = 0.0f;
    }
}

__global__ void k_group_sa2(const float* __restrict__ xyz1, const float* __restrict__ feat1,
                            const float* __restrict__ nx2, const int* __restrict__ nidx,
                            float* __restrict__ X, int N, int S) {
    int total = B_ * S * KNB_;
    for (int p = blockIdx.x * blockDim.x + threadIdx.x; p < total; p += gridDim.x * blockDim.x) {
        int b = p / (S * KNB_);
        int rem = p % (S * KNB_);
        int s = rem / KNB_;
        int id = nidx[p];
        const float* src = xyz1 + ((size_t)b * N + id) * 3;
        const float* ctr = nx2 + ((size_t)b * S + s) * 3;
        const float* f = feat1 + ((size_t)b * N + id) * 32;
        float* row = X + (size_t)p * 36;
        row[0] = src[0] - ctr[0]; row[1] = src[1] - ctr[1]; row[2] = src[2] - ctr[2];
#pragma unroll
        for (int c = 0; c < 32; ++c) row[3 + c] = f[c];
        row[35] = 0.0f;
    }
}

// ---------------------------------------------------------------------------
// Zero-pad weights [Cin,Cout] -> [Cin_pad,Cout]
// ---------------------------------------------------------------------------
__global__ void k_pad_weights(const float* __restrict__ W, float* __restrict__ Wp,
                              int Cin, int Cin_pad, int Cout) {
    int total = Cin_pad * Cout;
    for (int p = blockIdx.x * blockDim.x + threadIdx.x; p < total; p += gridDim.x * blockDim.x) {
        int r = p / Cout, c = p % Cout;
        Wp[p] = (r < Cin) ? W[r * Cout + c] : 0.0f;
    }
}

// ---------------------------------------------------------------------------
// WMMA matmul: Y[P,Cout] = X[P,Kd] * W[Kd,Cout].
// One wave computes a 16x16 tile, K stepped 4 (f32 WMMA) or 32 (f16 fallback).
// Layouts per CDNA5 ISA 7.12.2: A rows on lanes 0-15, K split on lane halves;
// B/D columns on lane&15, rows per VGPR (D: v + 8*hi).
// ---------------------------------------------------------------------------
__global__ void k_matmul_wmma(const float* __restrict__ X, const float* __restrict__ W,
                              float* __restrict__ Y, int P, int Kd, int Cout) {
    int wave = (blockIdx.x * blockDim.x + threadIdx.x) >> 5;
    int lane = threadIdx.x & 31;
    int ctiles = Cout >> 4;
    int ntiles = (P >> 4) * ctiles;
    if (wave >= ntiles) return;          // uniform per wave -> EXEC all 1s inside
    int rt = wave / ctiles, ct = wave % ctiles;
    int r = (rt << 4) + (lane & 15);
    int col = (ct << 4) + (lane & 15);
    int hi = lane >> 4;
    v8f acc = {};

#if __has_builtin(__builtin_amdgcn_wmma_f32_16x16x4_f32)
    for (int k0 = 0; k0 < Kd; k0 += 4) {
        int ka = k0 + 2 * hi;            // lanes 0-15: K=k0,k0+1 ; lanes 16-31: K=k0+2,k0+3
        v2f a, b;
        a.x = X[(size_t)r * Kd + ka];
        a.y = X[(size_t)r * Kd + ka + 1];
        b.x = W[(size_t)ka * Cout + col];
        b.y = W[(size_t)(ka + 1) * Cout + col];
        acc = __builtin_amdgcn_wmma_f32_16x16x4_f32(false, a, false, b,
                                                    (short)0, acc, false, false);
    }
#else
    for (int k0 = 0; k0 < Kd; k0 += 32) {
        v16h ah, bh;
#pragma unroll
        for (int h = 0; h < 16; ++h) {
            int v = h >> 1, pp = h & 1;
            int kl = (v < 4) ? (2 * v + pp) : (16 + 2 * (v - 4) + pp);
            int kk = k0 + kl + hi * 8;
            float xa = (kk < Kd) ? X[(size_t)r * Kd + kk] : 0.0f;
            float xb = (kk < Kd) ? W[(size_t)kk * Cout + col] : 0.0f;
            ah[h] = (_Float16)xa;
            bh[h] = (_Float16)xb;
        }
        acc = __builtin_amdgcn_wmma_f32_16x16x32_f16(false, ah, false, bh,
                                                     (short)0, acc, false, false);
    }
#endif

#pragma unroll
    for (int v = 0; v < 8; ++v) {
        int rr = (rt << 4) + v + hi * 8;
        Y[(size_t)rr * Cout + col] = acc[v];
    }
}

// ---------------------------------------------------------------------------
// InstanceNorm stats + normalize + ReLU (per batch, per channel over S*K)
// ---------------------------------------------------------------------------
__global__ void k_zero(float* __restrict__ p, int n) {
    for (int i = blockIdx.x * blockDim.x + threadIdx.x; i < n; i += gridDim.x * blockDim.x)
        p[i] = 0.0f;
}

__global__ void k_stats(const float* __restrict__ Y, float* __restrict__ stats,
                        int Ppb, int C) {
    int b = blockIdx.y;
    int t = threadIdx.x;                 // blockDim.x == 256, C | 256
    int c = t % C;
    int rg = t / C;
    int RPB = blockDim.x / C;
    float s = 0.0f, s2 = 0.0f;
    for (int r = blockIdx.x * RPB + rg; r < Ppb; r += gridDim.x * RPB) {
        float v = Y[((size_t)b * Ppb + r) * C + c];
        s += v; s2 += v * v;
    }
    atomicAdd(&stats[(b * C + c) * 2 + 0], s);
    atomicAdd(&stats[(b * C + c) * 2 + 1], s2);
}

__global__ void k_norm_relu(float* __restrict__ Y, const float* __restrict__ stats,
                            int Ppb, int C, int total) {
    float inv_n = 1.0f / (float)Ppb;
    for (int i = blockIdx.x * blockDim.x + threadIdx.x; i < total; i += gridDim.x * blockDim.x) {
        int b = i / (Ppb * C);
        int c = i % C;
        float mu = stats[(b * C + c) * 2 + 0] * inv_n;
        float var = stats[(b * C + c) * 2 + 1] * inv_n - mu * mu;
        var = fmaxf(var, 0.0f);
        float x = (Y[i] - mu) * rsqrtf(var + EPSN);
        Y[i] = fmaxf(x, 0.0f);
    }
}

// ---------------------------------------------------------------------------
// Max pool over K neighbors; output [B,S,C]
// ---------------------------------------------------------------------------
__global__ void k_maxpool(const float* __restrict__ Y, float* __restrict__ f,
                          int S, int C) {
    int total = B_ * S * C;
    for (int i = blockIdx.x * blockDim.x + threadIdx.x; i < total; i += gridDim.x * blockDim.x) {
        int c = i % C;
        int bs = i / C;                  // b*S + s
        float m = -3.4e38f;
        for (int k = 0; k < KNB_; ++k)
            m = fmaxf(m, Y[((size_t)bs * KNB_ + k) * C + c]);
        f[(size_t)bs * C + c] = m;
    }
}

__global__ void k_transpose_feat(const float* __restrict__ f, float* __restrict__ out,
                                 int S, int C) {
    int total = B_ * S * C;
    for (int i = blockIdx.x * blockDim.x + threadIdx.x; i < total; i += gridDim.x * blockDim.x) {
        int c = i % C;
        int bs = i / C;
        int b = bs / S, s = bs % S;
        out[((size_t)b * C + c) * S + s] = f[i];
    }
}

// ---------------------------------------------------------------------------
extern "C" void kernel_launch(void* const* d_in, const int* in_sizes, int n_in,
                              void* d_out, int out_size, void* d_ws, size_t ws_size,
                              hipStream_t stream) {
    const float* pc      = (const float*)d_in[0];
    const float* feature = (const float*)d_in[1];
    const float* sa1w1   = (const float*)d_in[2];
    const float* sa1w2   = (const float*)d_in[3];
    const float* sa1w3   = (const float*)d_in[4];
    const float* sa2w1   = (const float*)d_in[5];
    const float* sa2w2   = (const float*)d_in[6];
    const float* sa2w3   = (const float*)d_in[7];

    float* out      = (float*)d_out;
    float* out_pc   = out;                               // [4,3,16384]
    float* out_pcl1 = out + 196608;                      // [4,3,2048]
    float* out_pcl2 = out + 196608 + 24576;              // [4,3,1024]
    float* out_feat = out + 196608 + 24576 + 12288;      // [4,64,1024]
    int*   out_fidx1 = (int*)(out + 495616);             // [4,2048] int32 bits
    int*   out_fidx2 = (int*)(out + 503808);             // [4,1024] int32 bits

    char* w = (char*)d_ws;
    size_t off = 0;
    auto alloc = [&](size_t nbytes) -> char* {
        char* p = w + off;
        off = (off + nbytes + 255) & ~(size_t)255;
        return p;
    };
    float* xyz   = (float*)alloc((size_t)B_ * N1_ * 3 * 4);
    float* feat  = (float*)alloc((size_t)B_ * N1_ * 3 * 4);
    float* nx1   = (float*)alloc((size_t)B_ * S1_ * 3 * 4);
    float* feat1 = (float*)alloc((size_t)B_ * S1_ * 32 * 4);
    float* nx2   = (float*)alloc((size_t)B_ * S2_ * 3 * 4);
    float* feat2 = (float*)alloc((size_t)B_ * S2_ * 64 * 4);
    int*   fidx1 = (int*)alloc((size_t)B_ * S1_ * 4);
    int*   fidx2 = (int*)alloc((size_t)B_ * S2_ * 4);
    int*   nidx1 = (int*)alloc((size_t)B_ * S1_ * KNB_ * 4);
    int*   nidx2 = (int*)alloc((size_t)B_ * S2_ * KNB_ * 4);
    float* wpad  = (float*)alloc((size_t)64 * 64 * 4);
    float* stats = (float*)alloc((size_t)B_ * 64 * 2 * 4);
    float* bufA  = (float*)alloc((size_t)131072 * 36 * 4);   // groups (max of P1*8, P2*36)
    float* bufB  = (float*)alloc((size_t)262144 * 32 * 4);   // 32 MB ping
    float* bufC  = (float*)alloc((size_t)262144 * 32 * 4);   // 32 MB pong

    const int P1 = B_ * S1_ * KNB_;   // 262144
    const int P2 = B_ * S2_ * KNB_;   // 131072

    auto mm = [&](const float* X, const float* Wp, float* Y, int P, int Kd, int C) {
        int waves = (P / 16) * (C / 16);
        int blocks = (waves + 3) / 4;
        k_matmul_wmma<<<blocks, 128, 0, stream>>>(X, Wp, Y, P, Kd, C);
    };
    auto norm = [&](float* Y, int Ppb, int C) {
        k_zero<<<1, 512, 0, stream>>>(stats, B_ * 64 * 2);
        k_stats<<<dim3(32, B_), 256, 0, stream>>>(Y, stats, Ppb, C);
        int total = B_ * Ppb * C;
        k_norm_relu<<<2048, 256, 0, stream>>>(Y, stats, Ppb, C, total);
    };

    // ---- stage 0: layout ----
    k_transpose_in<<<256, 256, 0, stream>>>(pc, feature, xyz, feat, out_pc, B_, N1_);

    // ---- SA1 ----
    size_t smem1 = ((size_t)3 * N1_ + 80) * 4;   // xyz SoA + reduce scratch (<= 197 KB of 320 KB)
    k_fps<16><<<B_, 1024, smem1, stream>>>(xyz, fidx1, out_fidx1, N1_, S1_);
    k_gather_centers<<<64, 256, 0, stream>>>(xyz, fidx1, nx1, out_pcl1, N1_, S1_);
    k_knn<<<(B_ * S1_) / 128, 128, 0, stream>>>(nx1, xyz, nidx1, N1_, S1_);
    k_group_sa1<<<1024, 256, 0, stream>>>(xyz, feat, nx1, nidx1, bufA, N1_, S1_);

    k_pad_weights<<<16, 256, 0, stream>>>(sa1w1, wpad, 6, 8, 32);
    mm(bufA, wpad, bufB, P1, 8, 32);  norm(bufB, S1_ * KNB_, 32);
    k_pad_weights<<<16, 256, 0, stream>>>(sa1w2, wpad, 32, 32, 32);
    mm(bufB, wpad, bufC, P1, 32, 32); norm(bufC, S1_ * KNB_, 32);
    k_pad_weights<<<16, 256, 0, stream>>>(sa1w3, wpad, 32, 32, 32);
    mm(bufC, wpad, bufB, P1, 32, 32); norm(bufB, S1_ * KNB_, 32);
    k_maxpool<<<1024, 256, 0, stream>>>(bufB, feat1, S1_, 32);

    // ---- SA2 ----
    size_t smem2 = ((size_t)3 * S1_ + 80) * 4;
    k_fps<2><<<B_, 1024, smem2, stream>>>(nx1, fidx2, out_fidx2, S1_, S2_);
    k_gather_centers<<<32, 256, 0, stream>>>(nx1, fidx2, nx2, out_pcl2, S1_, S2_);
    k_knn<<<(B_ * S2_) / 128, 128, 0, stream>>>(nx2, nx1, nidx2, S1_, S2_);
    k_group_sa2<<<512, 256, 0, stream>>>(nx1, feat1, nx2, nidx2, bufA, S1_, S2_);

    k_pad_weights<<<16, 256, 0, stream>>>(sa2w1, wpad, 35, 36, 64);
    mm(bufA, wpad, bufB, P2, 36, 64); norm(bufB, S2_ * KNB_, 64);
    k_pad_weights<<<16, 256, 0, stream>>>(sa2w2, wpad, 64, 64, 64);
    mm(bufB, wpad, bufC, P2, 64, 64); norm(bufC, S2_ * KNB_, 64);
    k_pad_weights<<<16, 256, 0, stream>>>(sa2w3, wpad, 64, 64, 64);
    mm(bufC, wpad, bufB, P2, 64, 64); norm(bufB, S2_ * KNB_, 64);
    k_maxpool<<<1024, 256, 0, stream>>>(bufB, feat2, S2_, 64);
    k_transpose_feat<<<1024, 256, 0, stream>>>(feat2, out_feat, S2_, 64);
}
